// DiffTransformer_19877108645905
// MI455X (gfx1250) — matrix operand order repaired and checked by reading, not compiled, not run on hardware
//
#include <hip/hip_runtime.h>
#include <math.h>

// ---------------- model constants ----------------
#define DM     1024
#define SEQL   1024
#define BATCHN 2
#define ROWS   (BATCHN * SEQL)   // 2048
#define VOCAB  32000
#define DFF    4096
#define NHEAD  16

typedef __attribute__((ext_vector_type(16))) __bf16 v16bf;
typedef __attribute__((ext_vector_type(8)))  float  v8f;
typedef __attribute__((ext_vector_type(4)))  unsigned int u32x4;
typedef __attribute__((ext_vector_type(8)))  unsigned int u32x8;

union Frag {
  v16bf v;
  unsigned short s[16];
  unsigned int   u[8];
};

__device__ __forceinline__ unsigned short f2bf(float f) {
  unsigned u = __builtin_bit_cast(unsigned, f);
  u += 0x7FFFu + ((u >> 16) & 1u);           // round-to-nearest-even
  return (unsigned short)(u >> 16);
}

__device__ __forceinline__ v8f wmma_bf16(v16bf a, v16bf b, v8f c) {
  return __builtin_amdgcn_wmma_f32_16x16x32_bf16(false, a, false, b, (short)0, c,
                                                 false, false);
}

// ----------------------------------------------------------------------
// TDM: DMA a 2D fp32 tile (tile_d0 contiguous elems per row, tile_d1 rows,
// row stride = stride_elems) from global memory into LDS (row-major packed).
// D# built per CDNA5 ISA 08_async_tensor §8.3/8.4; groups 2/3 omitted (2D).
// Issued once per wave; tracked by TENSORcnt.
// ----------------------------------------------------------------------
__device__ __forceinline__ void tdm_load_2d(const void* lds_dst,
                                            const void* gsrc,
                                            unsigned tile_d0, unsigned tile_d1,
                                            unsigned stride_elems) {
  const unsigned long long ga = (unsigned long long)gsrc;
  const unsigned lds = (unsigned)(unsigned long long)lds_dst;
  u32x4 g0;
  g0[0] = 1u;                                             // count=1, user desc
  g0[1] = lds;                                            // lds_addr (bytes)
  g0[2] = (unsigned)ga;                                   // global_addr lo
  g0[3] = ((unsigned)(ga >> 32) & 0x01FFFFFFu) | (2u << 30);  // addr hi | type=2
  u32x8 g1;
  g1[0] = 2u << 16;                                       // data_size = 4 bytes
  g1[1] = (tile_d0 & 0xFFFFu) << 16;                      // tensor_dim0 lo16
  g1[2] = (tile_d0 >> 16) | ((tile_d1 & 0xFFFFu) << 16);  // td0 hi | tensor_dim1 lo
  g1[3] = (tile_d1 >> 16) | ((tile_d0 & 0xFFFFu) << 16);  // td1 hi | tile_dim0
  g1[4] = tile_d1 & 0xFFFFu;                              // tile_dim1 (tile_dim2=0)
  g1[5] = stride_elems;                                   // tensor_dim0_stride lo32
  g1[6] = 0u;                                             // stride hi | dim1_stride lo
  g1[7] = 0u;
  asm volatile("tensor_load_to_lds %0, %1" :: "s"(g0), "s"(g1) : "memory");
}

__device__ __forceinline__ float hmax16(float v) {
  v = fmaxf(v, __shfl_xor(v, 1, 32));
  v = fmaxf(v, __shfl_xor(v, 2, 32));
  v = fmaxf(v, __shfl_xor(v, 4, 32));
  v = fmaxf(v, __shfl_xor(v, 8, 32));
  return v;
}
__device__ __forceinline__ float hsum16(float v) {
  v += __shfl_xor(v, 1, 32);
  v += __shfl_xor(v, 2, 32);
  v += __shfl_xor(v, 4, 32);
  v += __shfl_xor(v, 8, 32);
  return v;
}
__device__ __forceinline__ float hsum32(float v) {
  v += __shfl_xor(v, 16, 32);
  return hsum16(v);
}

// ======================================================================
// Generic bf16 WMMA GEMM:  out[M,N] = epi(A[M,K] @ W[K,N] + bias)
// EPI: 0 plain, 1 silu, 2 multiply-by-aux, 3 add-aux(residual)
// Block: 256 thr (8 waves), WG tile 64x128, wave tile 32x32, K-step 32.
// fp32 tiles DMA'd to LDS by the Tensor Data Mover, double-buffered;
// fp32 -> bf16 conversion happens at fragment build time.
// ======================================================================
template <int EPI>
__global__ __launch_bounds__(256) void gemm_bf16_kernel(
    const float* __restrict__ A, const float* __restrict__ W,
    const float* __restrict__ bias, const float* __restrict__ aux,
    float* __restrict__ out, int M, int N, int K) {
  __shared__ __align__(16) float Asf[2][64 * 32];    // 2 x 8 KB
  __shared__ __align__(16) float Bsf[2][32 * 128];   // 2 x 16 KB

  const int tid  = threadIdx.x;
  const int lane = tid & 31;
  const int wave = tid >> 5;
  const int wm   = wave >> 2;        // 0..1
  const int wn   = wave & 3;         // 0..3
  const int m0   = blockIdx.y * 64;
  const int n0   = blockIdx.x * 128;
  const int mrow = lane & 15;
  const int hi   = lane >> 4;

  v8f acc[2][2] = {};

  const int nk = K / 32;
  if (wave == 0) {
    tdm_load_2d(&Asf[0][0], A + (long)m0 * K, 32, 64, (unsigned)K);
    tdm_load_2d(&Bsf[0][0], W + (long)0 * N + n0, 128, 32, (unsigned)N);
  }

  for (int step = 0; step < nk; ++step) {
    const int buf = step & 1;
    if (wave == 0) {
      if (step + 1 < nk) {
        const int kk = (step + 1) * 32;
        tdm_load_2d(&Asf[buf ^ 1][0], A + (long)m0 * K + kk, 32, 64,
                    (unsigned)K);
        tdm_load_2d(&Bsf[buf ^ 1][0], W + (long)kk * N + n0, 128, 32,
                    (unsigned)N);
        __builtin_amdgcn_s_wait_tensorcnt(2);  // current pair complete
      } else {
        __builtin_amdgcn_s_wait_tensorcnt(0);
      }
    }
    __syncthreads();

    Frag a[2], b[2];
#pragma unroll
    for (int mt = 0; mt < 2; ++mt) {
      const int row = wm * 32 + mt * 16 + mrow;
#pragma unroll
      for (int v = 0; v < 8; ++v) {
        const int kb = hi * 16 + 2 * v;
        const float2 f = *(const float2*)&Asf[buf][row * 32 + kb];
        a[mt].s[2 * v]     = f2bf(f.x);
        a[mt].s[2 * v + 1] = f2bf(f.y);
      }
    }
#pragma unroll
    for (int nt = 0; nt < 2; ++nt) {
      const int col = wn * 32 + nt * 16 + mrow;
#pragma unroll
      for (int v = 0; v < 8; ++v) {
        const int kr = hi * 16 + 2 * v;
        b[nt].s[2 * v]     = f2bf(Bsf[buf][kr * 128 + col]);
        b[nt].s[2 * v + 1] = f2bf(Bsf[buf][(kr + 1) * 128 + col]);
      }
    }
#pragma unroll
    for (int mt = 0; mt < 2; ++mt)
#pragma unroll
      for (int nt = 0; nt < 2; ++nt)
        acc[mt][nt] = wmma_bf16(a[mt].v, b[nt].v, acc[mt][nt]);
    __syncthreads();
  }

#pragma unroll
  for (int mt = 0; mt < 2; ++mt) {
#pragma unroll
    for (int nt = 0; nt < 2; ++nt) {
      const int col = n0 + wn * 32 + nt * 16 + mrow;
      const float bv = bias[col];
#pragma unroll
      for (int v = 0; v < 8; ++v) {
        const int row = m0 + wm * 32 + mt * 16 + v + 8 * hi;
        const long idx = (long)row * N + col;
        float val = acc[mt][nt][v] + bv;
        if constexpr (EPI == 1) val = val / (1.f + __expf(-val));   // silu
        if constexpr (EPI == 2) val *= aux[idx];                    // *gate
        if constexpr (EPI == 3) val += aux[idx];                    // +residual
        out[idx] = val;
      }
    }
  }
}

// ======================================================================
// Differential flash attention (WMMA scores + WMMA P@V), TDM-staged tiles.
// Block = 128 thr (4 waves) handles (b, head, 64-query block).
// ======================================================================
__device__ __forceinline__ void online_step(
    v8f s[2], int k0, int q0, int lane, int hi,
    float* m, float* l, v8f acc[4],
    unsigned short (*Pw)[32], const Frag bv[4]) {
  const int nl = lane & 15;
  // scale + causal mask
#pragma unroll
  for (int nt = 0; nt < 2; ++nt)
#pragma unroll
    for (int v = 0; v < 8; ++v) {
      const int key = k0 + nt * 16 + nl;
      const int row = q0 + v + 8 * hi;
      const float sc = s[nt][v] * 0.125f;
      s[nt][v] = (key <= row) ? sc : -3.0e38f;
    }
  float p0v[8], p1v[8];
#pragma unroll
  for (int v = 0; v < 8; ++v) {
    float rm = hmax16(fmaxf(s[0][v], s[1][v]));
    const float mn = fmaxf(m[v], rm);
    const float ef = __expf(m[v] - mn);
    const float p0 = __expf(s[0][v] - mn);
    const float p1 = __expf(s[1][v] - mn);
    l[v] = l[v] * ef + hsum16(p0 + p1);
    m[v] = mn;
#pragma unroll
    for (int t = 0; t < 4; ++t) acc[t][v] *= ef;
    p0v[v] = p0;
    p1v[v] = p1;
  }
  // transpose P (C-layout -> A-layout) through wave-private LDS
#pragma unroll
  for (int v = 0; v < 8; ++v) {
    Pw[v + 8 * hi][nl]      = f2bf(p0v[v]);
    Pw[v + 8 * hi][16 + nl] = f2bf(p1v[v]);
  }
  asm volatile("s_wait_dscnt 0" ::: "memory");
  Frag ap;
#pragma unroll
  for (int v = 0; v < 8; ++v) {
    const int kb = hi * 16 + 2 * v;
    ap.u[v] = *(const unsigned*)&Pw[nl][kb];
  }
#pragma unroll
  for (int t = 0; t < 4; ++t) acc[t] = wmma_bf16(ap.v, bv[t].v, acc[t]);
}

__global__ __launch_bounds__(128) void attn_kernel(
    const float* __restrict__ q, const float* __restrict__ kbuf,
    const float* __restrict__ vbuf, const float* __restrict__ hg,
    const float* __restrict__ lamPtr, float oneMinusLamInit,
    float* __restrict__ out) {
  __shared__ __align__(16) float Qf[64 * 64];                 // 16 KB
  __shared__ __align__(16) float Kf[32 * 32];                 //  4 KB
  __shared__ __align__(16) float Vf[32 * 64];                 //  8 KB
  __shared__ __align__(16) unsigned short Pls[4][16][32];     //  4 KB

  const int tid  = threadIdx.x;
  const int lane = tid & 31;
  const int wave = tid >> 5;
  const int bi   = blockIdx.x;
  const int qblk = bi & 15;
  const int h    = (bi >> 4) & 15;
  const int b    = bi >> 8;
  const int q0   = qblk * 64 + wave * 16;
  const int mrow = lane & 15;
  const int hi   = lane >> 4;
  const float lam = lamPtr[0];

  if (wave == 0) {
    tdm_load_2d(Qf, q + (long)(b * SEQL + qblk * 64) * DM + h * 64, 64, 64, DM);
    __builtin_amdgcn_s_wait_tensorcnt(0);
  }
  __syncthreads();

  Frag aq0, aq1;
#pragma unroll
  for (int v = 0; v < 8; ++v) {
    const int kb = hi * 16 + 2 * v;
    const float2 f0 = *(const float2*)&Qf[(wave * 16 + mrow) * 64 + kb];
    const float2 f1 = *(const float2*)&Qf[(wave * 16 + mrow) * 64 + 32 + kb];
    aq0.s[2 * v] = f2bf(f0.x); aq0.s[2 * v + 1] = f2bf(f0.y);
    aq1.s[2 * v] = f2bf(f1.x); aq1.s[2 * v + 1] = f2bf(f1.y);
  }

  float m0a[8], m1a[8], l0a[8], l1a[8];
#pragma unroll
  for (int v = 0; v < 8; ++v) { m0a[v] = m1a[v] = -3.0e38f; l0a[v] = l1a[v] = 0.f; }
  v8f acc0[4] = {}, acc1[4] = {};

  const int nchunk = (qblk + 1) * 2;
  for (int ch = 0; ch < nchunk; ++ch) {
    const int k0 = ch * 32;
    if (wave == 0) {
      tdm_load_2d(Kf, kbuf + (long)(b * SEQL + k0) * 512 + h * 32, 32, 32, 512);
      tdm_load_2d(Vf, vbuf + (long)(b * SEQL + k0) * 512 + (h >> 1) * 64, 64, 32,
                  512);
      __builtin_amdgcn_s_wait_tensorcnt(0);
    }
    __syncthreads();

    if (k0 < q0 + 16) {                 // wave-uniform causal skip
      Frag bk[2], bv[4];
#pragma unroll
      for (int nt = 0; nt < 2; ++nt)
#pragma unroll
        for (int v = 0; v < 8; ++v) {
          const float2 f = *(const float2*)&Kf[(nt * 16 + mrow) * 32 + hi * 16 + 2 * v];
          bk[nt].s[2 * v]     = f2bf(f.x);
          bk[nt].s[2 * v + 1] = f2bf(f.y);
        }
#pragma unroll
      for (int nt = 0; nt < 4; ++nt)
#pragma unroll
        for (int v = 0; v < 8; ++v) {
          const int kr = hi * 16 + 2 * v;
          bv[nt].s[2 * v]     = f2bf(Vf[kr * 64 + nt * 16 + mrow]);
          bv[nt].s[2 * v + 1] = f2bf(Vf[(kr + 1) * 64 + nt * 16 + mrow]);
        }
      const v8f z = {};
      v8f s0[2] = { wmma_bf16(aq0.v, bk[0].v, z), wmma_bf16(aq0.v, bk[1].v, z) };
      v8f s1[2] = { wmma_bf16(aq1.v, bk[0].v, z), wmma_bf16(aq1.v, bk[1].v, z) };
      online_step(s0, k0, q0, lane, hi, m0a, l0a, acc0, Pls[wave], bv);
      online_step(s1, k0, q0, lane, hi, m1a, l1a, acc1, Pls[wave], bv);
    }
    __syncthreads();
  }

  // diff combine + per-head rmsnorm(hg) * (1 - lam_init)
#pragma unroll
  for (int v = 0; v < 8; ++v) {
    const float inv0 = 1.f / l0a[v];
    const float inv1 = 1.f / l1a[v];
    float vals[4];
    float ss = 0.f;
#pragma unroll
    for (int t = 0; t < 4; ++t) {
      const float val = acc0[t][v] * inv0 - lam * (acc1[t][v] * inv1);
      vals[t] = val;
      ss += val * val;
    }
    ss = hsum16(ss);
    const float rs = rsqrtf(ss * (1.f / 64.f) + 1e-5f);
    const int row = q0 + v + 8 * hi;
#pragma unroll
    for (int t = 0; t < 4; ++t) {
      const int d = t * 16 + mrow;
      out[(long)(b * SEQL + row) * DM + h * 64 + d] =
          vals[t] * rs * hg[d] * oneMinusLamInit;
    }
  }
}

// ======================================================================
// small helper kernels
// ======================================================================
__global__ __launch_bounds__(256) void embed_kernel(
    const int* __restrict__ tok, const float* __restrict__ emb,
    float* __restrict__ x) {
  const long t = (long)blockIdx.x * 256 + threadIdx.x;   // ROWS*DM
  const int row = (int)(t >> 10), d = (int)(t & 1023);
  x[t] = emb[(long)tok[row] * DM + d];
}

__global__ __launch_bounds__(256) void rmsnorm_kernel(
    const float* __restrict__ x, const float* __restrict__ g,
    float* __restrict__ out) {
  const int wave = threadIdx.x >> 5, lane = threadIdx.x & 31;
  const long row = (long)blockIdx.x * 8 + wave;
  const float* xr = x + row * DM;
  float ss = 0.f;
  for (int j = lane; j < DM; j += 32) { const float v = xr[j]; ss += v * v; }
  ss = hsum32(ss);
  const float rs = rsqrtf(ss * (1.f / DM) + 1e-5f);
  float* orow = out + row * DM;
  for (int j = lane; j < DM; j += 32) orow[j] = xr[j] * rs * g[j];
}

__global__ __launch_bounds__(256) void rope_kernel(float* __restrict__ buf,
                                                   int nheads) {
  const long t = (long)blockIdx.x * 256 + threadIdx.x;  // B*S*nheads*16
  const int j = (int)(t & 15);
  const long r = t >> 4;
  const int h2 = (int)(r % nheads);
  const long bs = r / nheads;
  const int s = (int)(bs & (SEQL - 1));
  const float inv = powf(10000.f, -(float)j * (1.f / 16.f));
  const float ang = (float)s * inv;
  const float c = cosf(ang), sn = sinf(ang);
  const long base = (bs * nheads + h2) * 32;
  const float x1 = buf[base + j], x2 = buf[base + j + 16];
  buf[base + j]      = x1 * c - x2 * sn;
  buf[base + j + 16] = x1 * sn + x2 * c;
}

__global__ void lambda_kernel(const float* __restrict__ lk1,
                              const float* __restrict__ lq1,
                              const float* __restrict__ lk2,
                              const float* __restrict__ lq2,
                              float lam_init, float* __restrict__ lam) {
  const int lane = threadIdx.x;
  float d1 = lk1[lane] * lq1[lane];
  float d2 = lk2[lane] * lq2[lane];
  d1 = hsum32(d1);
  d2 = hsum32(d2);
  if (lane == 0) lam[0] = expf(d1) - expf(d2) + lam_init;
}

__global__ __launch_bounds__(256) void softmax_kernel(float* __restrict__ x,
                                                      int N) {
  __shared__ float red[256];
  const long row = blockIdx.x;
  float* p = x + row * N;
  const int tid = threadIdx.x;
  float mx = -3.0e38f;
  for (int j = tid; j < N; j += 256) mx = fmaxf(mx, p[j]);
  red[tid] = mx;
  __syncthreads();
  for (int s = 128; s > 0; s >>= 1) {
    if (tid < s) red[tid] = fmaxf(red[tid], red[tid + s]);
    __syncthreads();
  }
  mx = red[0];
  __syncthreads();
  float sum = 0.f;
  for (int j = tid; j < N; j += 256) sum += expf(p[j] - mx);
  red[tid] = sum;
  __syncthreads();
  for (int s = 128; s > 0; s >>= 1) {
    if (tid < s) red[tid] += red[tid + s];
    __syncthreads();
  }
  const float inv = 1.f / red[0];
  for (int j = tid; j < N; j += 256) p[j] = expf(p[j] - mx) * inv;
}

// ======================================================================
// host orchestration
// ======================================================================
extern "C" void kernel_launch(void* const* d_in, const int* in_sizes, int n_in,
                              void* d_out, int out_size, void* d_ws,
                              size_t ws_size, hipStream_t stream) {
  (void)in_sizes; (void)n_in; (void)out_size; (void)ws_size;

  const int* tokens  = (const int*)d_in[0];
  const float* emb   = (const float*)d_in[1];
  // params dict order per layer:
  // 0 qw 1 qb 2 kw 3 kb 4 vw 5 vb 6 ow 7 ob 8 lk1 9 lk2 10 lq1 11 lq2
  // 12 hg 13 n1 14 n2 15 gw 16 gb 17 uw 18 ub 19 dw 20 db
  const float* nf    = (const float*)d_in[2 + 2 * 21 + 0];
  const float* out_w = (const float*)d_in[2 + 2 * 21 + 1];
  const float* out_b = (const float*)d_in[2 + 2 * 21 + 2];

  float* ws   = (float*)d_ws;
  float* x    = ws;                        // 2048*1024
  float* xn   = ws + 2097152;              // 2048*1024
  float* qb_  = ws + 4194304;              // 2048*1024
  float* kb_  = ws + 6291456;              // 2048*512
  float* vb_  = ws + 7340032;              // 2048*512
  float* ao   = ws + 8388608;              // 2048*1024
  float* gate = ws + 10485760;             // 2048*4096 (h reused in place)
  float* lamp = ws + 18874368;             // 1

  embed_kernel<<<ROWS * DM / 256, 256, 0, stream>>>(tokens, emb, x);

  for (int layer = 0; layer < 2; ++layer) {
    const int base = 2 + layer * 21;
    auto in = [&](int o) { return (const float*)d_in[base + o]; };
    const float lam_init = 0.8f - 0.6f * expf(-0.3f * (float)layer);

    rmsnorm_kernel<<<ROWS / 8, 256, 0, stream>>>(x, in(13), xn);
    gemm_bf16_kernel<0><<<dim3(DM / 128, ROWS / 64), 256, 0, stream>>>(
        xn, in(0), in(1), nullptr, qb_, ROWS, DM, DM);
    gemm_bf16_kernel<0><<<dim3(512 / 128, ROWS / 64), 256, 0, stream>>>(
        xn, in(2), in(3), nullptr, kb_, ROWS, 512, DM);
    gemm_bf16_kernel<0><<<dim3(512 / 128, ROWS / 64), 256, 0, stream>>>(
        xn, in(4), in(5), nullptr, vb_, ROWS, 512, DM);
    rope_kernel<<<ROWS * 32 * 16 / 256, 256, 0, stream>>>(qb_, 32);
    rope_kernel<<<ROWS * 16 * 16 / 256, 256, 0, stream>>>(kb_, 16);
    lambda_kernel<<<1, 32, 0, stream>>>(in(8), in(10), in(9), in(11), lam_init,
                                        lamp);
    attn_kernel<<<BATCHN * NHEAD * (SEQL / 64), 128, 0, stream>>>(
        qb_, kb_, vb_, in(12), lamp, 1.f - lam_init, ao);
    gemm_bf16_kernel<3><<<dim3(DM / 128, ROWS / 64), 256, 0, stream>>>(
        ao, in(6), in(7), x, x, ROWS, DM, DM);

    rmsnorm_kernel<<<ROWS / 8, 256, 0, stream>>>(x, in(14), xn);
    gemm_bf16_kernel<1><<<dim3(DFF / 128, ROWS / 64), 256, 0, stream>>>(
        xn, in(15), in(16), nullptr, gate, ROWS, DFF, DM);
    gemm_bf16_kernel<2><<<dim3(DFF / 128, ROWS / 64), 256, 0, stream>>>(
        xn, in(17), in(18), gate, gate, ROWS, DFF, DM);
    gemm_bf16_kernel<3><<<dim3(DM / 128, ROWS / 64), 256, 0, stream>>>(
        gate, in(19), in(20), x, x, ROWS, DM, DFF);
  }

  rmsnorm_kernel<<<ROWS / 8, 256, 0, stream>>>(x, nf, xn);
  gemm_bf16_kernel<0><<<dim3(VOCAB / 128, ROWS / 64), 256, 0, stream>>>(
      xn, out_w, out_b, nullptr, (float*)d_out, ROWS, VOCAB, DM);
  softmax_kernel<<<ROWS, 256, 0, stream>>>((float*)d_out, VOCAB);
}
